// RNN_classifier_30614526886052
// MI455X (gfx1250) — matrix operand order repaired and checked by reading, not compiled, not run on hardware
//
#include <hip/hip_runtime.h>
#include <cstdint>
#include <cstddef>

// ---------------- problem sizes ----------------
#define Bz    128
#define Tz    64
#define HIMG  50
#define WIMG  60
#define EMBD  750          // conv embedding length
#define EPAD  768          // embedding padded to multiple of 32 (K dim)
#define OSZ   100          // Q/R/O size
#define NP    112          // 100 padded to 7 WMMA tiles of 16
#define KR    416          // 400 padded to multiple of 32
#define KQ    896          // 768 (padded emb) + 100 (R part) padded to 896
#define OUTC  10

typedef __attribute__((ext_vector_type(16))) _Float16 v16h;
typedef __attribute__((ext_vector_type(8)))  _Float16 v8h;
typedef __attribute__((ext_vector_type(8)))  float    v8f;

__constant__ float kAlphas[4] = {0.0f, 0.25f, 0.5f, 0.95f};

#define LEAKY(x) ((x) >= 0.0f ? (x) : 0.01f * (x))

// ---------------- workspace layout (bytes) ----------------
#define WS_E_OFF    0u
#define WS_E_BYTES  (8192u * EPAD * 2u)                 // 12,582,912
#define WS_RW_OFF   (WS_E_OFF + WS_E_BYTES)
#define WS_RW_BYTES (NP * KR * 2u)                      // 93,184
#define WS_QW_OFF   (WS_RW_OFF + WS_RW_BYTES)
#define WS_QW_BYTES (NP * KQ * 2u)                      // 200,704
#define WS_OW_OFF   (WS_QW_OFF + WS_QW_BYTES)
#define WS_OW_BYTES (NP * KR * 2u)                      // 93,184
#define WS_OUT_OFF  (WS_OW_OFF + WS_OW_BYTES)
#define WS_OUT_BYTES (Bz * Tz * OSZ * 4u)               // 3,276,800

// =====================================================================
// WMMA A-fragment loader, f16 source (global or LDS).
// Contiguous K order (matches B fragment):
// lane<16 : halves 0-15 -> K=kc..kc+15
// lane>=16: halves 0-15 -> K=kc+16..kc+31
// => two contiguous 16-byte loads per lane (ds_load_b128 / global_load_b128).
// =====================================================================
__device__ __forceinline__ v16h a_from_f16(const _Float16* rowp, int hi) {
  v8h lo = *(const v8h*)(rowp + hi * 16);
  v8h hh = *(const v8h*)(rowp + hi * 16 + 8);
  v16h a;
#pragma unroll
  for (int j = 0; j < 8; ++j) { a[j] = lo[j]; a[8 + j] = hh[j]; }
  return a;
}

// =====================================================================
// Kernel 1: fused conv1(3x3,s1,p1)+leaky -> conv2(3x3,s2,p1)+leaky embed
// One block per (b,t) image; conv1 streamed channel-by-channel via LDS.
// =====================================================================
__device__ __forceinline__ float c2sum(const float* sC, const float* w2t, int p) {
  int oi = p / 30, oj = p % 30;
  const float* base = sC + (2 * oi) * 62 + (2 * oj);
  return w2t[0]*base[0]   + w2t[1]*base[1]   + w2t[2]*base[2]
       + w2t[3]*base[62]  + w2t[4]*base[63]  + w2t[5]*base[64]
       + w2t[6]*base[124] + w2t[7]*base[125] + w2t[8]*base[126];
}

__global__ __launch_bounds__(256) void conv_embed_k(
    const float* __restrict__ In, const float* __restrict__ w1,
    const float* __restrict__ b1, const float* __restrict__ w2,
    const float* __restrict__ b2, _Float16* __restrict__ E) {
  __shared__ float sIn[52 * 62];
  __shared__ float sC[52 * 62];
  const int tid = threadIdx.x;
  const int blk = blockIdx.x;       // In is (B, T, 1, H, W): blk = b*64 + t
  const int b = blk >> 6;
  const int t = blk & 63;

  for (int i = tid; i < 52 * 62; i += 256) { sIn[i] = 0.f; sC[i] = 0.f; }
  __syncthreads();
  const float* img = In + (size_t)blk * (HIMG * WIMG);
  for (int i = tid; i < HIMG * WIMG; i += 256)
    sIn[(i / WIMG + 1) * 62 + (i % WIMG + 1)] = img[i];
  __syncthreads();

  float a0 = 0.f, a1 = 0.f, a2 = 0.f;
  for (int c = 0; c < 16; ++c) {
    float w1t[9];
#pragma unroll
    for (int u = 0; u < 9; ++u) w1t[u] = w1[c * 9 + u];
    float bb = b1[c];
    for (int idx = tid; idx < HIMG * WIMG; idx += 256) {
      int i = idx / WIMG, j = idx % WIMG;
      const float* bp = sIn + i * 62 + j;
      float s = bb
        + w1t[0]*bp[0]   + w1t[1]*bp[1]   + w1t[2]*bp[2]
        + w1t[3]*bp[62]  + w1t[4]*bp[63]  + w1t[5]*bp[64]
        + w1t[6]*bp[124] + w1t[7]*bp[125] + w1t[8]*bp[126];
      sC[(i + 1) * 62 + (j + 1)] = LEAKY(s);
    }
    __syncthreads();
    float w2t[9];
#pragma unroll
    for (int u = 0; u < 9; ++u) w2t[u] = w2[c * 9 + u];
    a0 += c2sum(sC, w2t, tid);
    a1 += c2sum(sC, w2t, tid + 256);
    if (tid < 238) a2 += c2sum(sC, w2t, tid + 512);
    __syncthreads();
  }
  float bb2 = b2[0];
  _Float16* erow = E + ((size_t)t * Bz + b) * EPAD;   // stored (t, b, 768)
  { float v = a0 + bb2; erow[tid]       = (_Float16)LEAKY(v); }
  { float v = a1 + bb2; erow[tid + 256] = (_Float16)LEAKY(v); }
  if (tid < 238) { float v = a2 + bb2; erow[tid + 512] = (_Float16)LEAKY(v); }
  if (tid < EPAD - EMBD) erow[EMBD + tid] = (_Float16)0.f;   // zero K-pad
}

// =====================================================================
// Kernel 2: weight prep — f32 [K][N] -> f16 [N][K] transposed + padded.
// Q weights re-indexed into padded-K space: [0,750)->emb, [768,868)->R.
// =====================================================================
__global__ __launch_bounds__(256) void prep_weights_k(
    const float* __restrict__ Rw, const float* __restrict__ Qw,
    const float* __restrict__ Ow, _Float16* __restrict__ RWt,
    _Float16* __restrict__ QWt, _Float16* __restrict__ OWt) {
  int idx = blockIdx.x * 256 + threadIdx.x;    // [0, NP*KQ)
  int n = idx / KQ;
  int kp = idx % KQ;
  if (n >= NP) return;
  float qv = 0.f;
  if (n < OSZ) {
    if (kp < EMBD) qv = Qw[(size_t)kp * OSZ + n];
    else if (kp >= EPAD && kp < EPAD + OSZ) qv = Qw[(size_t)(EMBD + kp - EPAD) * OSZ + n];
  }
  QWt[(size_t)n * KQ + kp] = (_Float16)qv;
  if (kp < KR) {
    float rv = (n < OSZ && kp < 400) ? Rw[(size_t)kp * OSZ + n] : 0.f;
    float ov = (n < OSZ && kp < 400) ? Ow[(size_t)kp * OSZ + n] : 0.f;
    RWt[(size_t)n * KR + kp] = (_Float16)rv;
    OWt[(size_t)n * KR + kp] = (_Float16)ov;
  }
}

// =====================================================================
// Kernel 3: persistent recurrence. 8 blocks x 256 threads; each block
// owns 16 batch rows for all 64 steps. Waves 0..6 each own one 16-wide
// N tile. f32 master H + f16 shadow H/Rt in LDS so every GEMM A-fragment
// is two raw ds_load_b128 (no f32->f16 cvts on the critical path).
// =====================================================================
__global__ __launch_bounds__(256) void rnn_core_k(
    const _Float16* __restrict__ E,  const _Float16* __restrict__ RWt,
    const _Float16* __restrict__ QWt, const _Float16* __restrict__ OWt,
    const float* __restrict__ Rb, const float* __restrict__ Qb,
    const float* __restrict__ Ob, float* __restrict__ OutBuf) {
  __shared__ __align__(16) float    sH [16 * KR];   // f32 master H (pad K zeroed)
  __shared__ __align__(16) _Float16 sHh[16 * KR];   // f16 shadow of H
  __shared__ __align__(16) float    sQ [16 * 128];  // Qt (f32, pad cols zeroed)
  __shared__ __align__(16) _Float16 sRh[16 * 128];  // Rt (f16, pad cols zeroed)
  __shared__ float sRb[OSZ], sQb[OSZ], sOb[OSZ];
  const int tid = threadIdx.x;
  for (int i = tid; i < 16 * KR; i += 256)  { sH[i] = 0.f; sHh[i] = (_Float16)0.f; }
  for (int i = tid; i < 16 * 128; i += 256) { sQ[i] = 0.f; sRh[i] = (_Float16)0.f; }
  if (tid < OSZ) { sRb[tid] = Rb[tid]; sQb[tid] = Qb[tid]; sOb[tid] = Ob[tid]; }
  __syncthreads();

  const int lane = tid & 31, wave = tid >> 5;
  const int row = lane & 15, hi = lane >> 4;
  const int n = wave * 16 + row;                  // output column this lane owns
  const int bBase = blockIdx.x * 16;

  for (int t = 0; t < Tz; ++t) {
    // ---- phase 1: Rt = tanh(H @ R_w + R_b)  -> sRh (f16) ----
    if (wave < 7) {
      v8f acc = {};
      const _Float16* hrow = sHh + row * KR;
      const _Float16* bcol = RWt + (size_t)n * KR + hi * 16;
#pragma unroll
      for (int kc = 0; kc < KR; kc += 32) {
        v16h a  = a_from_f16(hrow + kc, hi);
        v16h bm = *(const v16h*)(bcol + kc);
        acc = __builtin_amdgcn_wmma_f32_16x16x32_f16(false, a, false, bm,
                                                     (short)0, acc, false, false);
      }
      if (n < OSZ) {
        float bias = sRb[n];
#pragma unroll
        for (int r = 0; r < 8; ++r)
          sRh[(r + hi * 8) * 128 + n] = (_Float16)tanhf(acc[r] + bias);
      }
    }
    __syncthreads();

    // ---- phase 2: Qt = tanh([E_t, Rt] @ Q_w + Q_b) -> sQ (f32) ----
    if (wave < 7) {
      v8f qacc = {};
      const _Float16* erow = E + ((size_t)t * Bz + bBase + row) * EPAD;
      const _Float16* bcol = QWt + (size_t)n * KQ + hi * 16;
#pragma unroll
      for (int kc = 0; kc < EPAD; kc += 32) {      // embedding part (global f16)
        v16h a  = a_from_f16(erow + kc, hi);
        v16h bm = *(const v16h*)(bcol + kc);
        qacc = __builtin_amdgcn_wmma_f32_16x16x32_f16(false, a, false, bm,
                                                      (short)0, qacc, false, false);
      }
      const _Float16* xrow = sRh + row * 128;
#pragma unroll
      for (int kc = EPAD; kc < KQ; kc += 32) {     // Rt part (LDS f16)
        v16h a  = a_from_f16(xrow + (kc - EPAD), hi);
        v16h bm = *(const v16h*)(bcol + kc);
        qacc = __builtin_amdgcn_wmma_f32_16x16x32_f16(false, a, false, bm,
                                                      (short)0, qacc, false, false);
      }
      if (n < OSZ) {        // Qt goes to its own buffer: no barrier needed first
        float bias = sQb[n];
#pragma unroll
        for (int r = 0; r < 8; ++r)
          sQ[(r + hi * 8) * 128 + n] = tanhf(qacc[r] + bias);
      }
    }
    __syncthreads();

    // ---- phase 3: H = alpha*H + (1-alpha)*Qt  (f32 master + f16 shadow) ----
    for (int idx = tid; idx < 16 * 400; idx += 256) {
      int m = idx / 400, c = idx % 400;
      float alpha = kAlphas[c / 100];
      float q = sQ[m * 128 + (c % 100)];
      float nh = alpha * sH[m * KR + c] + (1.f - alpha) * q;
      sH[m * KR + c]  = nh;
      sHh[m * KR + c] = (_Float16)nh;
    }
    __syncthreads();

    // ---- phase 4: Out_t = tanh(H @ O_w + O_b) -> global ----
    if (wave < 7) {
      v8f acc = {};
      const _Float16* hrow = sHh + row * KR;
      const _Float16* bcol = OWt + (size_t)n * KR + hi * 16;
#pragma unroll
      for (int kc = 0; kc < KR; kc += 32) {
        v16h a  = a_from_f16(hrow + kc, hi);
        v16h bm = *(const v16h*)(bcol + kc);
        acc = __builtin_amdgcn_wmma_f32_16x16x32_f16(false, a, false, bm,
                                                     (short)0, acc, false, false);
      }
      if (n < OSZ) {
        float bias = sOb[n];
#pragma unroll
        for (int r = 0; r < 8; ++r)
          OutBuf[(size_t)(bBase + r + hi * 8) * (Tz * OSZ) + t * OSZ + n] =
              tanhf(acc[r] + bias);
      }
    }
    // no barrier needed here: next phase-1 only reads sHh (no writer until
    // next phase-3, which is fenced by two barriers) and writes sRh (whose
    // readers in this step all finished before the post-phase-2 barrier).
  }
}

// =====================================================================
// Kernel 4: logits = Out @ Out_w + Out_b, softmax. One block per batch row.
// =====================================================================
__global__ __launch_bounds__(256) void classify_k(
    const float* __restrict__ OutBuf, const float* __restrict__ Ww,
    const float* __restrict__ Wb, float* __restrict__ out) {
  __shared__ float sred[OUTC][256];
  const int tid = threadIdx.x;
  float acc[OUTC];
#pragma unroll
  for (int c = 0; c < OUTC; ++c) acc[c] = 0.f;
  const float* rowp = OutBuf + (size_t)blockIdx.x * (Tz * OSZ);
  for (int k = tid; k < Tz * OSZ; k += 256) {
    float x = rowp[k];
    const float* wrow = Ww + (size_t)k * OUTC;
#pragma unroll
    for (int c = 0; c < OUTC; ++c) acc[c] += x * wrow[c];
  }
#pragma unroll
  for (int c = 0; c < OUTC; ++c) sred[c][tid] = acc[c];
  __syncthreads();
  for (int s = 128; s > 0; s >>= 1) {
    if (tid < s) {
#pragma unroll
      for (int c = 0; c < OUTC; ++c) sred[c][tid] += sred[c][tid + s];
    }
    __syncthreads();
  }
  if (tid == 0) {
    float lg[OUTC], mx = -1e30f;
#pragma unroll
    for (int c = 0; c < OUTC; ++c) { lg[c] = sred[c][0] + Wb[c]; mx = fmaxf(mx, lg[c]); }
    float sum = 0.f;
#pragma unroll
    for (int c = 0; c < OUTC; ++c) { lg[c] = expf(lg[c] - mx); sum += lg[c]; }
    float inv = 1.f / sum;
#pragma unroll
    for (int c = 0; c < OUTC; ++c) out[blockIdx.x * OUTC + c] = lg[c] * inv;
  }
}

// =====================================================================
extern "C" void kernel_launch(void* const* d_in, const int* in_sizes, int n_in,
                              void* d_out, int out_size, void* d_ws, size_t ws_size,
                              hipStream_t stream) {
  (void)in_sizes; (void)n_in; (void)out_size; (void)ws_size;
  const float* In   = (const float*)d_in[0];
  const float* c1w  = (const float*)d_in[1];
  const float* c1b  = (const float*)d_in[2];
  const float* c2w  = (const float*)d_in[3];
  const float* c2b  = (const float*)d_in[4];
  const float* Rw   = (const float*)d_in[5];
  const float* Rb   = (const float*)d_in[6];
  const float* Qw   = (const float*)d_in[7];
  const float* Qb   = (const float*)d_in[8];
  const float* Ow   = (const float*)d_in[9];
  const float* Ob   = (const float*)d_in[10];
  const float* Outw = (const float*)d_in[11];
  const float* Outb = (const float*)d_in[12];
  float* out = (float*)d_out;

  char* ws = (char*)d_ws;
  _Float16* E    = (_Float16*)(ws + WS_E_OFF);
  _Float16* RWt  = (_Float16*)(ws + WS_RW_OFF);
  _Float16* QWt  = (_Float16*)(ws + WS_QW_OFF);
  _Float16* OWt  = (_Float16*)(ws + WS_OW_OFF);
  float*    OutB = (float*)(ws + WS_OUT_OFF);

  conv_embed_k<<<Bz * Tz, 256, 0, stream>>>(In, c1w, c1b, c2w, c2b, E);
  prep_weights_k<<<(NP * KQ + 255) / 256, 256, 0, stream>>>(Rw, Qw, Ow, RWt, QWt, OWt);
  rnn_core_k<<<Bz / 16, 256, 0, stream>>>(E, RWt, QWt, OWt, Rb, Qb, Ob, OutB);
  classify_k<<<Bz, 256, 0, stream>>>(OutB, Outw, Outb, out);
}
